// TraditionalGNN_61787399520423
// MI455X (gfx1250) — matrix operand order repaired and checked by reading, not compiled, not run on hardware
//
#include <hip/hip_runtime.h>
#include <hip/hip_bf16.h>

typedef __attribute__((ext_vector_type(16))) _Float16 v16h;
typedef __attribute__((ext_vector_type(8)))  _Float16 v8h;
typedef __attribute__((ext_vector_type(8)))  float    v8f;

// ---------------------------------------------------------------- utilities
__global__ void k_f32_to_f16(const float* __restrict__ in, _Float16* __restrict__ out, int n) {
  int i = blockIdx.x * blockDim.x + threadIdx.x;
  if (i < n) out[i] = (_Float16)in[i];
}

// W[K,Nc] f32 -> Wt[Nc,K] f16 (transposed so GEMM B-fragments are contiguous)
__global__ void k_w_to_f16_t(const float* __restrict__ W, _Float16* __restrict__ Wt,
                             int Kd, int Nc) {
  int i = blockIdx.x * blockDim.x + threadIdx.x;
  if (i >= Kd * Nc) return;
  int k = i / Nc, n = i - k * Nc;
  Wt[(size_t)n * Kd + k] = (_Float16)W[i];
}

__global__ void k_fill(float* __restrict__ p, float v, int n) {
  int i = blockIdx.x * blockDim.x + threadIdx.x;
  if (i < n) p[i] = v;
}

// ------------------------------------------------------------- WMMA GEMM
// C[M,Nc](f32) = A[M,128](f16, row-major) x B[128,Nc](f16, given as Wt[Nc,128]).
// One wave computes a 16x64 output block: A fragment loaded once per k-step,
// reused across 4 WMMAs. All fragment loads are 16B (global_load_b128).
__device__ __forceinline__ v16h ld_frag16(const _Float16* p) {
  v8h lo = *(const v8h*)p;
  v8h hi = *(const v8h*)(p + 8);
  v16h r;
#pragma unroll
  for (int i = 0; i < 8; ++i) { r[i] = lo[i]; r[i + 8] = hi[i]; }
  return r;
}

__global__ void k_gemm_wmma(const _Float16* __restrict__ A, const _Float16* __restrict__ Wt,
                            float* __restrict__ C, int Nc, int total_waves) {
  constexpr int K = 128;
  int wave = blockIdx.x * (blockDim.x >> 5) + (threadIdx.x >> 5);
  if (wave >= total_waves) return;          // wave-uniform: EXEC stays all-1s
  int lane = threadIdx.x & 31;
  int l  = lane & 15;
  int hi = lane >> 4;                       // 0: lanes 0-15, 1: lanes 16-31
  int wavecols = Nc >> 6;                   // 64 output columns per wave
  int tm = (wave / wavecols) << 4;
  int tn = (wave % wavecols) << 6;

  const _Float16* arow = A + (size_t)(tm + l) * K;
  const _Float16* br0  = Wt + (size_t)(tn +  0 + l) * K;
  const _Float16* br1  = Wt + (size_t)(tn + 16 + l) * K;
  const _Float16* br2  = Wt + (size_t)(tn + 32 + l) * K;
  const _Float16* br3  = Wt + (size_t)(tn + 48 + l) * K;

  v8f acc0 = {}, acc1 = {}, acc2 = {}, acc3 = {};
#pragma unroll
  for (int k0 = 0; k0 < K; k0 += 32) {
    // A 16x32 f16: lane(l,hi): elem j -> K=k0+hi*16+j (contiguous, matches B)
    v8h alo = *(const v8h*)(arow + k0 + (hi << 4));
    v8h ahi = *(const v8h*)(arow + k0 + 8 + (hi << 4));
    v16h a;
#pragma unroll
    for (int i = 0; i < 8; ++i) { a[i] = alo[i]; a[i + 8] = ahi[i]; }

    // B 32x16 f16: lane(l,hi) col N; elem j -> K=k0+hi*16+j (contiguous in Wt)
    v16h b0 = ld_frag16(br0 + k0 + (hi << 4));
    v16h b1 = ld_frag16(br1 + k0 + (hi << 4));
    v16h b2 = ld_frag16(br2 + k0 + (hi << 4));
    v16h b3 = ld_frag16(br3 + k0 + (hi << 4));

    acc0 = __builtin_amdgcn_wmma_f32_16x16x32_f16(false, a, false, b0, (short)0, acc0, false, false);
    acc1 = __builtin_amdgcn_wmma_f32_16x16x32_f16(false, a, false, b1, (short)0, acc1, false, false);
    acc2 = __builtin_amdgcn_wmma_f32_16x16x32_f16(false, a, false, b2, (short)0, acc2, false, false);
    acc3 = __builtin_amdgcn_wmma_f32_16x16x32_f16(false, a, false, b3, (short)0, acc3, false, false);
  }

  // C/D 16x16 f32: VGPR r -> row tm + r + hi*8, col = tn + tile*16 + l
#pragma unroll
  for (int r = 0; r < 8; ++r) {
    float* crow = C + (size_t)(tm + r + (hi << 3)) * Nc + tn + l;
    crow[0]  = acc0[r];
    crow[16] = acc1[r];
    crow[32] = acc2[r];
    crow[48] = acc3[r];
  }
}

// ------------------------------------------------------- attention pieces
__global__ void k_node_alpha(const float* __restrict__ H,
                             const float* __restrict__ a_src,
                             const float* __restrict__ a_dst,
                             float* __restrict__ S, float* __restrict__ D,
                             int N, int heads, int dh) {
  int idx = blockIdx.x * blockDim.x + threadIdx.x;
  if (idx >= N * heads) return;
  int n = idx / heads, h = idx - n * heads;
  const float* hr = H + (size_t)n * heads * dh + h * dh;
  float s = 0.f, d = 0.f;
  for (int i = 0; i < dh; ++i) {
    s += hr[i] * a_src[h * dh + i];
    d += hr[i] * a_dst[h * dh + i];
  }
  S[idx] = s;
  D[idx] = d;
}

__device__ inline void atomicMaxFloat(float* addr, float v) {
  if (v >= 0.f) atomicMax((int*)addr, __float_as_int(v));
  else          atomicMin((unsigned int*)addr, __float_as_uint(v));
}

__device__ inline void edge_endpoints(const int* __restrict__ ei, int e, int E,
                                      int& src, int& dst) {
  if (e < E) { src = ei[e]; dst = ei[E + e]; }
  else       { src = dst = e - E; }          // appended self-loops
}

// pass 1: logit = leaky_relu(S[src]+D[dst]); store logit; segment max over dst
__global__ void k_edge_pass1(const int* __restrict__ ei, int E, int Etot,
                             const float* __restrict__ S, const float* __restrict__ D,
                             float* __restrict__ Mx, float* __restrict__ EL, int heads) {
  int idx = blockIdx.x * blockDim.x + threadIdx.x;
  if (idx >= Etot * heads) return;
  int e = idx / heads, h = idx - e * heads;
  int src, dst; edge_endpoints(ei, e, E, src, dst);
  float x = S[src * heads + h] + D[dst * heads + h];
  float lg = x > 0.f ? x : 0.2f * x;
  EL[idx] = lg;
  atomicMaxFloat(Mx + dst * heads + h, lg);
}

// pass 2: a = exp(logit - max); store a; segment sum over dst
__global__ void k_edge_pass2(const int* __restrict__ ei, int E, int Etot,
                             const float* __restrict__ Mx,
                             float* __restrict__ EL, float* __restrict__ Den, int heads) {
  int idx = blockIdx.x * blockDim.x + threadIdx.x;
  if (idx >= Etot * heads) return;
  int e = idx / heads, h = idx - e * heads;
  int src, dst; edge_endpoints(ei, e, E, src, dst);
  float a = __expf(EL[idx] - Mx[dst * heads + h]);
  EL[idx] = a;
  atomicAdd(&Den[dst * heads + h], a);
}

// Den -> 1/(Den+eps)   (400K rcp instead of 109M divides downstream)
__global__ void k_recip(float* __restrict__ Den, int n) {
  int i = blockIdx.x * blockDim.x + threadIdx.x;
  if (i < n) Den[i] = 1.0f / (Den[i] + 1e-16f);
}

// EL -> normalized attention weight alpha = EL * Rden[dst,h]
__global__ void k_edge_alpha(const int* __restrict__ ei, int E, int Etot,
                             const float* __restrict__ Rden, float* __restrict__ EL,
                             int heads) {
  int idx = blockIdx.x * blockDim.x + threadIdx.x;
  if (idx >= Etot * heads) return;
  int e = idx / heads, h = idx - e * heads;
  int src, dst; edge_endpoints(ei, e, E, src, dst);
  (void)src;
  EL[idx] *= Rden[dst * heads + h];
}

// aggregate: OUT[dst, f..f+3] += alpha * H[src, f..f+3]
// one thread per (edge, 4-feature chunk); lF4 = log2(F/4), ldh = log2(dh)
__global__ void k_edge_aggregate(const int* __restrict__ ei, int E, int Etot,
                                 const float* __restrict__ EL,
                                 const float* __restrict__ H, float* __restrict__ OUT,
                                 int heads, int ldh, int lF4) {
  int idx = blockIdx.x * blockDim.x + threadIdx.x;
  if (idx >= (Etot << lF4)) return;
  int e = idx >> lF4;
  int f = (idx & ((1 << lF4) - 1)) << 2;     // 4 consecutive features
  int h = f >> ldh;                          // dh multiple of 4 -> same head
  int F = 4 << lF4;
  int src, dst; edge_endpoints(ei, e, E, src, dst);
  float alpha = EL[(size_t)e * heads + h];
  const float4 hv = *(const float4*)(H + (size_t)src * F + f);
  float* o = OUT + (size_t)dst * F + f;
  atomicAdd(o + 0, alpha * hv.x);
  atomicAdd(o + 1, alpha * hv.y);
  atomicAdd(o + 2, alpha * hv.z);
  atomicAdd(o + 3, alpha * hv.w);
}

// finalize: + bias, relu, convert to f16 as next layer's GEMM input
__global__ void k_bias_relu_half(const float* __restrict__ acc, const float* __restrict__ b,
                                 _Float16* __restrict__ out, int total, int F) {
  int i = blockIdx.x * blockDim.x + threadIdx.x;
  if (i >= total) return;
  float v = acc[i] + b[i & (F - 1)];
  out[i] = (_Float16)(v > 0.f ? v : 0.f);
}

__global__ void k_bias_out(const float* __restrict__ acc, const float* __restrict__ b,
                           float* __restrict__ out, int total, int F) {
  int i = blockIdx.x * blockDim.x + threadIdx.x;
  if (i >= total) return;
  out[i] = acc[i] + b[i & (F - 1)];
}

// ---------------------------------------------------------------- launcher
extern "C" void kernel_launch(void* const* d_in, const int* in_sizes, int n_in,
                              void* d_out, int out_size, void* d_ws, size_t ws_size,
                              hipStream_t stream) {
  const float* x  = (const float*)d_in[0];
  const int*   ei = (const int*)d_in[1];
  const float* W[3]  = {(const float*)d_in[2], (const float*)d_in[6],  (const float*)d_in[10]};
  const float* As[3] = {(const float*)d_in[3], (const float*)d_in[7],  (const float*)d_in[11]};
  const float* Ad[3] = {(const float*)d_in[4], (const float*)d_in[8],  (const float*)d_in[12]};
  const float* Bb[3] = {(const float*)d_in[5], (const float*)d_in[9],  (const float*)d_in[13]};

  const int N    = in_sizes[0] / 128;   // 50000 (divisible by 16)
  const int E    = in_sizes[1] / 2;     // 800000
  const int Etot = E + N;               // + self loops

  char* ws = (char*)d_ws;
  size_t off = 0;
  auto alloc = [&](size_t bytes) -> char* {
    char* p = ws + off;
    off += (bytes + 255) & ~(size_t)255;
    return p;
  };
  _Float16* Ah  = (_Float16*)alloc((size_t)N * 128 * sizeof(_Float16));
  _Float16* Wt  = (_Float16*)alloc((size_t)128 * 128 * sizeof(_Float16));
  float*    Hm  = (float*)alloc((size_t)N * 128 * sizeof(float));
  float*    S   = (float*)alloc((size_t)N * 8 * sizeof(float));
  float*    Dd  = (float*)alloc((size_t)N * 8 * sizeof(float));
  float*    Mx  = (float*)alloc((size_t)N * 8 * sizeof(float));
  float*    Den = (float*)alloc((size_t)N * 8 * sizeof(float));
  float*    EL  = (float*)alloc((size_t)Etot * 8 * sizeof(float));
  float*    OUT = (float*)alloc((size_t)N * 128 * sizeof(float));
  (void)ws_size; (void)n_in; (void)out_size;

  auto cdiv = [](int a, int b) { return (a + b - 1) / b; };

  k_f32_to_f16<<<cdiv(N * 128, 256), 256, 0, stream>>>(x, Ah, N * 128);

  for (int layer = 0; layer < 3; ++layer) {
    const int heads = (layer == 2) ? 1 : 8;
    const int dh    = (layer == 2) ? 64 : 16;
    const int ldh   = (layer == 2) ? 6 : 4;   // log2(dh)
    const int F     = heads * dh;             // 128,128,64
    const int lF4   = (layer == 2) ? 4 : 5;   // log2(F/4)

    // weight -> transposed f16
    k_w_to_f16_t<<<cdiv(128 * F, 256), 256, 0, stream>>>(W[layer], Wt, 128, F);

    // projection via WMMA (one wave per 16x64 output block, 4 waves/block)
    const int waves = (N / 16) * (F / 64);
    k_gemm_wmma<<<cdiv(waves, 4), 128, 0, stream>>>(Ah, Wt, Hm, F, waves);

    // per-node attention halves
    const int na = N * heads;
    k_node_alpha<<<cdiv(na, 256), 256, 0, stream>>>(Hm, As[layer], Ad[layer], S, Dd, N, heads, dh);

    // init accumulators
    k_fill<<<cdiv(na, 256), 256, 0, stream>>>(Mx, -__builtin_inff(), na);
    k_fill<<<cdiv(na, 256), 256, 0, stream>>>(Den, 0.f, na);
    k_fill<<<cdiv(N * F, 256), 256, 0, stream>>>(OUT, 0.f, N * F);

    // segment softmax
    const int eh = Etot * heads;
    k_edge_pass1<<<cdiv(eh, 256), 256, 0, stream>>>(ei, E, Etot, S, Dd, Mx, EL, heads);
    k_edge_pass2<<<cdiv(eh, 256), 256, 0, stream>>>(ei, E, Etot, Mx, EL, Den, heads);
    k_recip<<<cdiv(na, 256), 256, 0, stream>>>(Den, na);
    k_edge_alpha<<<cdiv(eh, 256), 256, 0, stream>>>(ei, E, Etot, Den, EL, heads);

    // weighted aggregation (vectorized gather, atomic scatter into L2-resident OUT)
    const int aggt = Etot << lF4;
    k_edge_aggregate<<<cdiv(aggt, 256), 256, 0, stream>>>(ei, E, Etot, EL, Hm, OUT,
                                                          heads, ldh, lF4);

    // bias (+ relu + f16 re-pack) / final output
    if (layer < 2)
      k_bias_relu_half<<<cdiv(N * F, 256), 256, 0, stream>>>(OUT, Bb[layer], Ah, N * F, F);
    else
      k_bias_out<<<cdiv(N * F, 256), 256, 0, stream>>>(OUT, Bb[layer], (float*)d_out, N * F, F);
  }
}
